// Int8DotGeneralOp_58428735095556
// MI455X (gfx1250) — compile-verified, matrix operand order checked by reading, not run on hardware
//
#include <hip/hip_runtime.h>

typedef __attribute__((ext_vector_type(8))) int v8i;

constexpr int Mdim = 8192;
constexpr int Kdim = 4096;
constexpr int Ndim = 4096;
constexpr float QMAX = 127.0f;

// ---------------------------------------------------------------------------
// CDNA5 async global->LDS copy (GLOBAL_LOAD_ASYNC_TO_LDS_B128, ASYNCcnt).
// IOFFSET is added to BOTH the LDS and global addresses per ISA pseudocode.
// ---------------------------------------------------------------------------
template <int OFF>
__device__ __forceinline__ void async_copy_b128(unsigned lds_off,
                                                const signed char* gaddr) {
  asm volatile("global_load_async_to_lds_b128 %0, %1, off offset:%2"
               :
               : "v"(lds_off), "v"(gaddr), "n"(OFF)
               : "memory");
}

__device__ __forceinline__ void wait_asynccnt_le4() {
  asm volatile("s_wait_asynccnt 4" ::: "memory");
}
__device__ __forceinline__ void wait_asynccnt_0() {
  asm volatile("s_wait_asynccnt 0" ::: "memory");
}

// ---------------------------------------------------------------------------
// 1) lhs: per-row abs-max -> scale, quantize to int8 (row-major, K contiguous)
// ---------------------------------------------------------------------------
__global__ __launch_bounds__(256) void quant_lhs_kernel(
    const float* __restrict__ lhs, signed char* __restrict__ lhs_q,
    float* __restrict__ s_lhs) {
  const int row = blockIdx.x;
  const int t = threadIdx.x;
  const float* rp = lhs + (size_t)row * Kdim;

  float4 v[4];
  float amax = 0.0f;
#pragma unroll
  for (int i = 0; i < 4; ++i) {
    v[i] = ((const float4*)rp)[t + i * 256];
    amax = fmaxf(amax, fmaxf(fmaxf(fabsf(v[i].x), fabsf(v[i].y)),
                             fmaxf(fabsf(v[i].z), fabsf(v[i].w))));
  }

  __shared__ float red[256];
  red[t] = amax;
  __syncthreads();
  for (int s = 128; s > 0; s >>= 1) {
    if (t < s) red[t] = fmaxf(red[t], red[t + s]);
    __syncthreads();
  }
  const float am = red[0];
  const float scale = am > 0.0f ? am * (1.0f / QMAX) : 1.0f;
  const float inv = 1.0f / scale;
  if (t == 0) s_lhs[row] = scale;

  unsigned int* qp = (unsigned int*)(lhs_q + (size_t)row * Kdim);
#pragma unroll
  for (int i = 0; i < 4; ++i) {
    int q0 = (int)rintf(fminf(fmaxf(v[i].x * inv, -QMAX), QMAX));
    int q1 = (int)rintf(fminf(fmaxf(v[i].y * inv, -QMAX), QMAX));
    int q2 = (int)rintf(fminf(fmaxf(v[i].z * inv, -QMAX), QMAX));
    int q3 = (int)rintf(fminf(fmaxf(v[i].w * inv, -QMAX), QMAX));
    unsigned int packed = (q0 & 255u) | ((q1 & 255u) << 8) |
                          ((q2 & 255u) << 16) | ((unsigned)(q3 & 255u) << 24);
    qp[t + i * 256] = packed;
  }
}

// ---------------------------------------------------------------------------
// 2) rhs per-column abs-max, phase 1: partial over K chunks (grid-parallel)
// ---------------------------------------------------------------------------
__global__ __launch_bounds__(256) void rhs_absmax_partial_kernel(
    const float* __restrict__ rhs, float* __restrict__ partial) {
  const int col = blockIdx.x * 256 + threadIdx.x;
  const int kchunk = Kdim / 16;
  const float* p = rhs + (size_t)(blockIdx.y * kchunk) * Ndim + col;
  float amax = 0.0f;
#pragma unroll 4
  for (int k = 0; k < kchunk; ++k) amax = fmaxf(amax, fabsf(p[(size_t)k * Ndim]));
  partial[blockIdx.y * Ndim + col] = amax;
}

__global__ __launch_bounds__(256) void rhs_scale_kernel(
    const float* __restrict__ partial, float* __restrict__ s_rhs) {
  const int col = blockIdx.x * 256 + threadIdx.x;
  float amax = 0.0f;
#pragma unroll
  for (int i = 0; i < 16; ++i) amax = fmaxf(amax, partial[i * Ndim + col]);
  s_rhs[col] = amax > 0.0f ? amax * (1.0f / QMAX) : 1.0f;
}

// ---------------------------------------------------------------------------
// 3) rhs: quantize + transpose to rhs_qT [N,K] via 64x64 LDS tile
// ---------------------------------------------------------------------------
__global__ __launch_bounds__(256) void rhs_quant_transpose_kernel(
    const float* __restrict__ rhs, const float* __restrict__ s_rhs,
    signed char* __restrict__ rhs_qT) {
  constexpr int LDT = 80;
  __shared__ signed char tile[64 * LDT];
  const int n0 = blockIdx.x * 64;
  const int k0 = blockIdx.y * 64;
  const int t = threadIdx.x;
  const int nl = t & 63;
  const int kq = t >> 6;  // 0..3

  const float inv = 1.0f / s_rhs[n0 + nl];
#pragma unroll
  for (int r = 0; r < 16; ++r) {
    const int kl = kq * 16 + r;
    const float x = rhs[(size_t)(k0 + kl) * Ndim + (n0 + nl)];
    const float q = fminf(fmaxf(rintf(x * inv), -QMAX), QMAX);
    tile[nl * LDT + kl] = (signed char)(int)q;
  }
  __syncthreads();

  const int orow = t >> 2;
  const int oc = (t & 3) * 16;
  const int4 val = *(const int4*)&tile[orow * LDT + oc];
  *(int4*)(rhs_qT + (size_t)(n0 + orow) * Kdim + k0 + oc) = val;
}

// ---------------------------------------------------------------------------
// 4) int8 WMMA GEMM, double-buffered LDS fed by async global->LDS copies.
//    C[128x128] per block, BK=64, 8 waves of 32(M)x64(N); fused dequant.
// ---------------------------------------------------------------------------
__global__ __launch_bounds__(256) void int8_wmma_gemm_kernel(
    const signed char* __restrict__ A,   // lhs_q [M,K]
    const signed char* __restrict__ Bt,  // rhs_qT [N,K]
    const float* __restrict__ s_lhs, const float* __restrict__ s_rhs,
    float* __restrict__ out) {
  constexpr int LDT = 80;          // padded pitch (16B-aligned, de-conflicted)
  constexpr int TILE = 128 * LDT;  // one 128x64 int8 tile (padded)
  __shared__ signed char As[2 * TILE];
  __shared__ signed char Bs[2 * TILE];

  const int t = threadIdx.x;
  const int lane = t & 31;
  const int wave = t >> 5;    // 0..7 (wave32)
  const int wm = wave & 3;    // 4 waves along M -> 32 rows each
  const int wn = wave >> 2;   // 2 waves along N -> 64 cols each
  const int m0 = blockIdx.y * 128;
  const int n0 = blockIdx.x * 128;

  // cooperative staging: thread -> (row 0..127, 32B half of the 64B K-slab)
  const int ldRow = t >> 1;
  const int ldHalf = (t & 1) * 32;
  const signed char* aSrc = A + (size_t)(m0 + ldRow) * Kdim + ldHalf;
  const signed char* bSrc = Bt + (size_t)(n0 + ldRow) * Kdim + ldHalf;
  const unsigned aDst = (unsigned)(size_t)(void*)&As[ldRow * LDT + ldHalf];
  const unsigned bDst = (unsigned)(size_t)(void*)&Bs[ldRow * LDT + ldHalf];

  auto stage = [&](int buf, int k0) {
    async_copy_b128<0>(aDst + buf * TILE, aSrc + k0);
    async_copy_b128<16>(aDst + buf * TILE, aSrc + k0);
    async_copy_b128<0>(bDst + buf * TILE, bSrc + k0);
    async_copy_b128<16>(bDst + buf * TILE, bSrc + k0);
  };

  const v8i zero = {0, 0, 0, 0, 0, 0, 0, 0};
  v8i acc[2][4];
#pragma unroll
  for (int mi = 0; mi < 2; ++mi)
#pragma unroll
    for (int ni = 0; ni < 4; ++ni) acc[mi][ni] = zero;

  const int lrow = lane & 15;      // row/col within the 16x16 tile
  const int lh = (lane >> 4) * 8;  // K-half select per ISA 8-bit layout

  constexpr int NK = Kdim / 64;
  stage(0, 0);  // prologue: fill buffer 0

  for (int it = 0; it < NK; ++it) {
    const int cur = it & 1;
    if (it + 1 < NK) {
      stage(cur ^ 1, (it + 1) * 64);  // overlap next copy with this compute
      if (it + 2 < NK) {              // warm L2 two steps ahead
        __builtin_prefetch(aSrc + (it + 2) * 64, 0, 1);
        __builtin_prefetch(bSrc + (it + 2) * 64, 0, 1);
      }
      wait_asynccnt_le4();  // 4 newest = next stage; current stage landed
    } else {
      wait_asynccnt_0();
    }
    __syncthreads();

    const signed char* Abuf = &As[cur * TILE];
    const signed char* Bbuf = &Bs[cur * TILE];

    // fragment loads per ISA 8-bit 16x64 layout:
    //   lanes 0-15: K = {0-7,16-23,32-39,48-55}; lanes 16-31: +8
    v8i afrag[2], bfrag[4];
#pragma unroll
    for (int mi = 0; mi < 2; ++mi) {
      const signed char* ap = &Abuf[(wm * 32 + mi * 16 + lrow) * LDT];
#pragma unroll
      for (int j = 0; j < 4; ++j) {
        const unsigned long long d =
            *(const unsigned long long*)(ap + j * 16 + lh);
        afrag[mi][2 * j] = (int)(unsigned int)d;
        afrag[mi][2 * j + 1] = (int)(unsigned int)(d >> 32);
      }
    }
#pragma unroll
    for (int ni = 0; ni < 4; ++ni) {
      const signed char* bp = &Bbuf[(wn * 64 + ni * 16 + lrow) * LDT];
#pragma unroll
      for (int j = 0; j < 4; ++j) {
        const unsigned long long d =
            *(const unsigned long long*)(bp + j * 16 + lh);
        bfrag[ni][2 * j] = (int)(unsigned int)d;
        bfrag[ni][2 * j + 1] = (int)(unsigned int)(d >> 32);
      }
    }

#pragma unroll
    for (int mi = 0; mi < 2; ++mi)
#pragma unroll
      for (int ni = 0; ni < 4; ++ni)
        acc[mi][ni] = __builtin_amdgcn_wmma_i32_16x16x64_iu8(
            /*sgn_a=*/true, afrag[mi], /*sgn_b=*/true, bfrag[ni],
            acc[mi][ni], /*reuse_a=*/false, /*reuse_b=*/false);

    __syncthreads();  // all waves done with buf[cur] before it is re-filled
  }

  // dequant epilogue: C layout lane L -> N=L%16; VGPR v -> M = 8*(L/16)+v
#pragma unroll
  for (int mi = 0; mi < 2; ++mi) {
    const int rowBase = m0 + wm * 32 + mi * 16 + (lane >> 4) * 8;
    const float4 sl0 = *(const float4*)&s_lhs[rowBase];      // rowBase % 8 == 0
    const float4 sl1 = *(const float4*)&s_lhs[rowBase + 4];
    const float sl[8] = {sl0.x, sl0.y, sl0.z, sl0.w, sl1.x, sl1.y, sl1.z, sl1.w};
#pragma unroll
    for (int ni = 0; ni < 4; ++ni) {
      const int col = n0 + wn * 64 + ni * 16 + (lane & 15);
      const float sn = s_rhs[col];
#pragma unroll
      for (int v = 0; v < 8; ++v) {
        out[(size_t)(rowBase + v) * Ndim + col] =
            (float)acc[mi][ni][v] * sl[v] * sn;
      }
    }
  }
}

// ---------------------------------------------------------------------------
extern "C" void kernel_launch(void* const* d_in, const int* in_sizes, int n_in,
                              void* d_out, int out_size, void* d_ws,
                              size_t ws_size, hipStream_t stream) {
  (void)in_sizes; (void)n_in; (void)out_size; (void)ws_size;
  const float* lhs = (const float*)d_in[0];
  const float* rhs = (const float*)d_in[1];
  float* out = (float*)d_out;

  char* ws = (char*)d_ws;
  signed char* lhs_q = (signed char*)ws;                          // M*K int8
  signed char* rhs_qT = (signed char*)(ws + (size_t)Mdim * Kdim); // N*K int8
  float* s_lhs = (float*)(ws + (size_t)Mdim * Kdim + (size_t)Ndim * Kdim);
  float* s_rhs = s_lhs + Mdim;
  float* partial = s_rhs + Ndim;  // 16 * N floats

  quant_lhs_kernel<<<Mdim, 256, 0, stream>>>(lhs, lhs_q, s_lhs);
  rhs_absmax_partial_kernel<<<dim3(Ndim / 256, 16), 256, 0, stream>>>(rhs, partial);
  rhs_scale_kernel<<<Ndim / 256, 256, 0, stream>>>(partial, s_rhs);
  rhs_quant_transpose_kernel<<<dim3(Ndim / 64, Kdim / 64), 256, 0, stream>>>(
      rhs, s_rhs, rhs_qT);
  int8_wmma_gemm_kernel<<<dim3(Ndim / 128, Mdim / 128), 256, 0, stream>>>(
      lhs_q, rhs_qT, s_lhs, s_rhs, out);
}
